// GPT2FullySharedAttention_55181739819319
// MI455X (gfx1250) — matrix operand order repaired and checked by reading, not compiled, not run on hardware
//
#include <hip/hip_runtime.h>
#include <hip/hip_bf16.h>

typedef __attribute__((ext_vector_type(16))) _Float16 v16h;
typedef __attribute__((ext_vector_type(8)))  _Float16 v8h;
typedef __attribute__((ext_vector_type(4)))  _Float16 v4h;
typedef __attribute__((ext_vector_type(8)))  float    v8f;

#define N_HEAD 16
#define BB     2
#define TT     2048
#define CC     1024
#define DD     64
#define MTOT   (BB*TT)   // 4096

union HFrag { v16h v; v8h h2[2]; };

// ---------------------------------------------------------------------------
// f32 -> f16 convert (vectorized x4)
// ---------------------------------------------------------------------------
__global__ __launch_bounds__(256) void cvt_f32_f16(const float4* __restrict__ s,
                                                   v4h* __restrict__ d, int n4) {
  int i = blockIdx.x * blockDim.x + threadIdx.x;
  if (i < n4) {
    float4 f = s[i];
    v4h h;
    h.x = (_Float16)f.x; h.y = (_Float16)f.y;
    h.z = (_Float16)f.z; h.w = (_Float16)f.w;
    d[i] = h;
  }
}

// ---------------------------------------------------------------------------
// WMMA GEMM: out = A[M,K] * W[N,K]^T + bias[N]
// MODE 0: out f16 as QK  [B,H,T,D]   (row m = b*T+t, col n = h*64+d)
// MODE 1: out f16 as V^T [B,H,D,T]
// MODE 2: out f32 row-major [M,N] (final projection, to d_out)
// Each wave computes a 32x64 strip (B fragments reused for 2 M-tiles);
// block = 8 waves -> 256 rows x 64 cols.
// ---------------------------------------------------------------------------
template<int MODE>
__global__ __launch_bounds__(256) void gemm_xwT(
    const _Float16* __restrict__ A,    // [M][K]
    const _Float16* __restrict__ W,    // [N][K]
    const float*    __restrict__ bias, // [N]
    _Float16* __restrict__ out16,
    float*    __restrict__ out32,
    int M, int N, int K)
{
  const int lane = threadIdx.x & 31;
  const int wave = threadIdx.x >> 5;
  const int rlo  = lane & 15;
  const int hi   = lane >> 4;

  const int m0 = blockIdx.y * 256 + wave * 32;
  const int n0 = blockIdx.x * 64;

  const _Float16* arow0 = A + (size_t)(m0 + rlo) * K + hi * 16;
  const _Float16* arow1 = arow0 + (size_t)16 * K;

  v8f acc[2][4] = {};

  for (int k0 = 0; k0 < K; k0 += 32) {
    HFrag a[2];
    a[0].h2[0] = *(const v8h*)(arow0 + k0);
    a[0].h2[1] = *(const v8h*)(arow0 + k0 + 8);
    a[1].h2[0] = *(const v8h*)(arow1 + k0);
    a[1].h2[1] = *(const v8h*)(arow1 + k0 + 8);
#pragma unroll
    for (int nt = 0; nt < 4; ++nt) {
      const _Float16* wrow = W + (size_t)(n0 + nt * 16 + rlo) * K + k0 + hi * 16;
      v16h b = *(const v16h*)wrow;             // 16 contiguous K at fixed n
#pragma unroll
      for (int mt = 0; mt < 2; ++mt)
        acc[mt][nt] = __builtin_amdgcn_wmma_f32_16x16x32_f16(
            false, a[mt].v, false, b, (short)0, acc[mt][nt], false, false);
    }
  }

#pragma unroll
  for (int mt = 0; mt < 2; ++mt)
#pragma unroll
    for (int nt = 0; nt < 4; ++nt) {
      const int n = n0 + nt * 16 + rlo;
      const float bv = bias[n];
#pragma unroll
      for (int r = 0; r < 8; ++r) {
        const int m = m0 + mt * 16 + r + 8 * hi;
        const float val = acc[mt][nt][r] + bv;
        if (MODE == 0) {
          const int b = m >> 11, t = m & (TT - 1);       // T = 2048
          const int h = n >> 6,  d = n & (DD - 1);
          out16[(((size_t)(b * N_HEAD + h) * TT + t) << 6) + d] = (_Float16)val;
        } else if (MODE == 1) {
          const int b = m >> 11, t = m & (TT - 1);
          const int h = n >> 6,  d = n & (DD - 1);
          out16[((size_t)(b * N_HEAD + h) * DD + d) * TT + t] = (_Float16)val;
        } else {
          out32[(size_t)m * CC + n] = val;
        }
      }
    }
}

// ---------------------------------------------------------------------------
// One key-tile step of the flash attention loop.
// MASK     : apply causal mask (only for the diagonal tile, kbase == q0)
// LOADNEXT : double-buffer the K fragments for tile kbase+32
// ---------------------------------------------------------------------------
template<bool MASK, bool LOADNEXT>
__device__ __attribute__((always_inline)) inline void attn_tile(
    const _Float16* __restrict__ Kb, const _Float16* __restrict__ Vb,
    const HFrag (&qa)[2][2], v16h (&kf)[2][2], v8f (&oacc)[2][4],
    float& m_old, float& l_sum, int kbase, int q0,
    int lane, int rlo, int hi,
    float (*Ssc)[33], _Float16 (*Pp)[32], float* rowf)
{
  const float kscale = 0.125f * 1.44269504088896f;  // 1/sqrt(D) * log2(e)

  // ---- S = Q K^T for 2x2 subtiles -----------------------------------------
  v8f sacc[2][2];
#pragma unroll
  for (int mt = 0; mt < 2; ++mt)
#pragma unroll
    for (int j = 0; j < 2; ++j) {
      v8f s = {};
      s = __builtin_amdgcn_wmma_f32_16x16x32_f16(false, qa[mt][0].v, false,
                                                 kf[j][0], (short)0, s, false, false);
      s = __builtin_amdgcn_wmma_f32_16x16x32_f16(false, qa[mt][1].v, false,
                                                 kf[j][1], (short)0, s, false, false);
      sacc[mt][j] = s;
    }

  // ---- issue V loads early (consumed after softmax) -----------------------
  v16h vf[4];
#pragma unroll
  for (int dt = 0; dt < 4; ++dt) {
    const _Float16* vrow = Vb + (size_t)(dt * 16 + rlo) * TT + kbase + hi * 16;
    vf[dt] = *(const v16h*)vrow;
  }

  // ---- prefetch K fragments for next tile (double buffer) ----------------
  v16h kn[2][2];
  if (LOADNEXT) {
#pragma unroll
    for (int j = 0; j < 2; ++j) {
      const _Float16* krow = Kb + (size_t)(kbase + 32 + j * 16 + rlo) * DD + hi * 16;
      kn[j][0] = *(const v16h*)(krow);
      kn[j][1] = *(const v16h*)(krow + 32);
    }
    // cache prefetch two tiles ahead
    int kpf = kbase + 64; if (kpf > TT - 32) kpf = 0;
    __builtin_prefetch(Kb + (size_t)(kpf + lane) * DD, 0, 1);      // 32 K rows
    __builtin_prefetch(Vb + (size_t)lane * TT + kpf, 0, 1);        // V rows 0..31
    __builtin_prefetch(Vb + (size_t)(lane + 32) * TT + kpf, 0, 1); // V rows 32..63
  }

  // ---- scores -> LDS (mask only on the diagonal tile) ---------------------
#pragma unroll
  for (int mt = 0; mt < 2; ++mt)
#pragma unroll
    for (int j = 0; j < 2; ++j)
#pragma unroll
      for (int r = 0; r < 8; ++r) {
        const int row = mt * 16 + r + 8 * hi;          // query row in tile
        float v = sacc[mt][j][r] * kscale;
        if (MASK) {
          const int kcol = kbase + j * 16 + rlo;
          v = (kcol <= q0 + row) ? v : -1e30f;
        }
        Ssc[row][j * 16 + rlo] = v;
      }
  asm volatile("s_wait_dscnt 0" ::: "memory");

  // ---- online softmax: lane L owns row L ----------------------------------
  {
    float mx = -1e30f;
#pragma unroll
    for (int c = 0; c < 32; ++c) mx = fmaxf(mx, Ssc[lane][c]);
    const float mnew = fmaxf(m_old, mx);
    float lt = 0.f;
#pragma unroll
    for (int c = 0; c < 32; ++c) {
      const float p = __builtin_amdgcn_exp2f(Ssc[lane][c] - mnew);
      lt += p;
      Pp[lane][c] = (_Float16)p;
    }
    const float sc = __builtin_amdgcn_exp2f(m_old - mnew);
    l_sum = l_sum * sc + lt;
    m_old = mnew;
    rowf[lane] = sc;
  }
  asm volatile("s_wait_dscnt 0" ::: "memory");

  // ---- rescale O accumulators ---------------------------------------------
#pragma unroll
  for (int mt = 0; mt < 2; ++mt) {
    float scr[8];
#pragma unroll
    for (int r = 0; r < 8; ++r) scr[r] = rowf[mt * 16 + r + 8 * hi];
#pragma unroll
    for (int dt = 0; dt < 4; ++dt)
#pragma unroll
      for (int r = 0; r < 8; ++r) oacc[mt][dt][r] *= scr[r];
  }

  // ---- O += P @ V ----------------------------------------------------------
#pragma unroll
  for (int mt = 0; mt < 2; ++mt) {
    HFrag pa;
    pa.h2[0] = *(const v8h*)(&Pp[mt * 16 + rlo][hi * 16]);
    pa.h2[1] = *(const v8h*)(&Pp[mt * 16 + rlo][hi * 16 + 8]);
#pragma unroll
    for (int dt = 0; dt < 4; ++dt)
      oacc[mt][dt] = __builtin_amdgcn_wmma_f32_16x16x32_f16(
          false, pa.v, false, vf[dt], (short)0, oacc[mt][dt], false, false);
  }

  // ---- rotate K double buffer ---------------------------------------------
  if (LOADNEXT) {
#pragma unroll
    for (int j = 0; j < 2; ++j) {
      kf[j][0] = kn[j][0];
      kf[j][1] = kn[j][1];
    }
  }
}

// ---------------------------------------------------------------------------
// Flash attention, causal, shared Q==K projection.
// One wave (block of 32) owns a 32-query tile (2 M-tiles) of one (b,h).
// Key loop split: q0/32 fully-unmasked tiles (no compares in the hot loop)
// followed by exactly one masked diagonal tile.
// QK: [B,H,T,D] f16 ; Vt: [B,H,D,T] f16 ; Y: [B,T,C] f16 (heads merged)
// ---------------------------------------------------------------------------
__global__ __launch_bounds__(32) void attn_flash(
    const _Float16* __restrict__ QK,
    const _Float16* __restrict__ Vt,
    _Float16* __restrict__ Y)
{
  __shared__ float    Ssc[32][33];   // scores (pad: stride 33, gcd(33,64)=1)
  __shared__ _Float16 Pp[32][32];    // probabilities (f16, A-frag source)
  __shared__ float    rowf[32];      // per-row rescale / inv-sum broadcast

  const int lane = threadIdx.x & 31;
  const int rlo  = lane & 15;
  const int hi   = lane >> 4;

  const int q0 = blockIdx.x * 32;
  const int h  = blockIdx.y;
  const int b  = blockIdx.z;

  const _Float16* Kb = QK + (size_t)(b * N_HEAD + h) * TT * DD;
  const _Float16* Vb = Vt + (size_t)(b * N_HEAD + h) * DD * TT;

  // Q fragments: 2 M-tiles x 2 d-chunks
  HFrag qa[2][2];
#pragma unroll
  for (int mt = 0; mt < 2; ++mt) {
    const _Float16* qrow = Kb + (size_t)(q0 + mt * 16 + rlo) * DD + hi * 16;
    qa[mt][0].h2[0] = *(const v8h*)(qrow);
    qa[mt][0].h2[1] = *(const v8h*)(qrow + 8);
    qa[mt][1].h2[0] = *(const v8h*)(qrow + 32);
    qa[mt][1].h2[1] = *(const v8h*)(qrow + 40);
  }

  v8f oacc[2][4] = {};
  float m_old = -__builtin_inff();   // every lane owns score-row `lane`
  float l_sum = 0.f;

  // preload K fragments for tile 0: [j = key subtile][c = d chunk]
  v16h kf[2][2];
#pragma unroll
  for (int j = 0; j < 2; ++j) {
    const _Float16* krow = Kb + (size_t)(j * 16 + rlo) * DD + hi * 16;
    kf[j][0] = *(const v16h*)(krow);
    kf[j][1] = *(const v16h*)(krow + 32);
  }

  // fully-unmasked tiles: keys [0, q0)
  const int nfull = q0 >> 5;
  for (int kt = 0; kt < nfull; ++kt)
    attn_tile<false, true>(Kb, Vb, qa, kf, oacc, m_old, l_sum,
                           kt << 5, q0, lane, rlo, hi, Ssc, Pp, rowf);

  // single masked diagonal tile: keys [q0, q0+32)
  attn_tile<true, false>(Kb, Vb, qa, kf, oacc, m_old, l_sum,
                         q0, q0, lane, rlo, hi, Ssc, Pp, rowf);

  // ---- finalize: divide by row sums, store Y[b, q, h*64 + d] --------------
  rowf[lane] = 1.0f / l_sum;
  asm volatile("s_wait_dscnt 0" ::: "memory");

  _Float16* ybase = Y + ((size_t)b * TT + q0) * CC + h * DD;
#pragma unroll
  for (int mt = 0; mt < 2; ++mt) {
    float inv[8];
#pragma unroll
    for (int r = 0; r < 8; ++r) inv[r] = rowf[mt * 16 + r + 8 * hi];
#pragma unroll
    for (int dt = 0; dt < 4; ++dt)
#pragma unroll
      for (int r = 0; r < 8; ++r) {
        const int m = mt * 16 + r + 8 * hi;
        ybase[(size_t)m * CC + dt * 16 + rlo] = (_Float16)(oacc[mt][dt][r] * inv[r]);
      }
  }
}

// ---------------------------------------------------------------------------
extern "C" void kernel_launch(void* const* d_in, const int* in_sizes, int n_in,
                              void* d_out, int out_size, void* d_ws, size_t ws_size,
                              hipStream_t stream) {
  const float* X   = (const float*)d_in[0];
  const float* Wqk = (const float*)d_in[1];
  const float* bqk = (const float*)d_in[2];
  const float* Wv  = (const float*)d_in[3];
  const float* bv  = (const float*)d_in[4];
  const float* Wo  = (const float*)d_in[5];
  const float* bo  = (const float*)d_in[6];
  float* out = (float*)d_out;

  const size_t NX = (size_t)MTOT * CC;   // 4M
  const size_t NW = (size_t)CC * CC;     // 1M

  _Float16* ws    = (_Float16*)d_ws;
  _Float16* X16   = ws;
  _Float16* Wqk16 = X16   + NX;
  _Float16* Wv16  = Wqk16 + NW;
  _Float16* Wo16  = Wv16  + NW;
  _Float16* QK16  = Wo16  + NW;
  _Float16* Vt16  = QK16  + NX;
  _Float16* Y16   = Vt16  + NX;

  // 1) downcast inputs to f16
  cvt_f32_f16<<<(int)(NX / 4 / 256), 256, 0, stream>>>((const float4*)X,  (v4h*)X16,   (int)(NX / 4));
  cvt_f32_f16<<<(int)(NW / 4 / 256), 256, 0, stream>>>((const float4*)Wqk,(v4h*)Wqk16, (int)(NW / 4));
  cvt_f32_f16<<<(int)(NW / 4 / 256), 256, 0, stream>>>((const float4*)Wv, (v4h*)Wv16,  (int)(NW / 4));
  cvt_f32_f16<<<(int)(NW / 4 / 256), 256, 0, stream>>>((const float4*)Wo, (v4h*)Wo16,  (int)(NW / 4));

  dim3 ggrid(CC / 64, MTOT / 256);       // (16, 16)

  // 2) QK projection -> [B,H,T,D] f16
  gemm_xwT<0><<<ggrid, 256, 0, stream>>>(X16, Wqk16, bqk, QK16, nullptr, MTOT, CC, CC);
  // 3) V projection -> [B,H,D,T] f16 (transposed for PV fragment loads)
  gemm_xwT<1><<<ggrid, 256, 0, stream>>>(X16, Wv16, bv, Vt16, nullptr, MTOT, CC, CC);
  // 4) causal flash attention with shared Q==K (32 queries per wave)
  attn_flash<<<dim3(TT / 32, N_HEAD, BB), 32, 0, stream>>>(QK16, Vt16, Y16);
  // 5) output projection -> f32 d_out
  gemm_xwT<2><<<ggrid, 256, 0, stream>>>(Y16, Wo16, bo, nullptr, out, MTOT, CC, CC);
}
